// SUTransformer_17386027615048
// MI455X (gfx1250) — matrix operand order repaired and checked by reading, not compiled, not run on hardware
//
#include <hip/hip_runtime.h>
#include <hip/hip_bf16.h>
#include <math.h>
#include <stdint.h>

// ---------------------------------------------------------------------------
// Problem constants (fixed by the reference)
// ---------------------------------------------------------------------------
#define BB 2
#define SS 2048
#define DD 2048
#define HH 16
#define DNN 128
#define DRR 64
#define DVV 128
#define LATT 512
#define DFFF 8192
#define MM (BB * SS)              // 4096 token rows
#define HDQ (DNN + DRR)           // 192
#define ATTN_SCALE 0.07216878364870323f   // 1/sqrt(192)

typedef __bf16 v16bf __attribute__((ext_vector_type(16)));
typedef float  v8f   __attribute__((ext_vector_type(8)));

union FragB16 {
    uint4 q[2];
    v16bf v;
    unsigned short u[16];
};

__device__ __forceinline__ unsigned short f2bf(float f) {
    union { float f; unsigned u; } c; c.f = f;
    unsigned r = c.u + 0x7FFFu + ((c.u >> 16) & 1u);
    return (unsigned short)(r >> 16);
}
__device__ __forceinline__ float bf2f(unsigned short h) {
    union { unsigned u; float f; } c; c.u = ((unsigned)h) << 16; return c.f;
}
__device__ __forceinline__ v8f zero8() {
    v8f z;
#pragma unroll
    for (int i = 0; i < 8; ++i) z[i] = 0.0f;
    return z;
}
__device__ __forceinline__ v8f wmma_bf16(const FragB16& a, const FragB16& b, v8f c) {
    return __builtin_amdgcn_wmma_f32_16x16x32_bf16(false, a.v, false, b.v,
                                                   (short)0, c, false, false);
}

// ---------------------------------------------------------------------------
// CDNA5 async global->LDS copy (ASYNCcnt path) via inline asm.
// vdst = VGPR holding LDS byte address, vaddr = 64-bit global address pair.
// Generic __shared__ pointers carry the LDS offset in their low 32 bits
// (flat aperture).
// ---------------------------------------------------------------------------
#ifndef __has_builtin
#define __has_builtin(x) 0
#endif

__device__ __forceinline__ void async_cp16(const void* g, void* l) {
    asm volatile("global_load_async_to_lds_b128 %0, %1, off"
                 :: "v"((unsigned)(uintptr_t)l), "v"(g)
                 : "memory");
}
template <int N>
__device__ __forceinline__ void wait_async() {
#if __has_builtin(__builtin_amdgcn_s_wait_asynccnt)
    __builtin_amdgcn_s_wait_asynccnt(N);
#else
    asm volatile("s_wait_asynccnt %0" :: "n"(N) : "memory");
#endif
}

// ---------------------------------------------------------------------------
// Weight convert + transpose: W f32 [K][N] -> Wt bf16 [N][K]
// ---------------------------------------------------------------------------
__global__ __launch_bounds__(256) void k_wt(const float* __restrict__ W,
                                            unsigned short* __restrict__ Wt,
                                            int K, int N) {
    __shared__ float t[32][33];
    const int nb = blockIdx.x * 32, kb = blockIdx.y * 32;
    const int tx = threadIdx.x & 31, ty = threadIdx.x >> 5;   // 32 x 8
#pragma unroll
    for (int i = 0; i < 32; i += 8)
        t[ty + i][tx] = W[(size_t)(kb + ty + i) * N + nb + tx];
    __syncthreads();
#pragma unroll
    for (int i = 0; i < 32; i += 8)
        Wt[(size_t)(nb + ty + i) * K + kb + tx] = f2bf(t[tx][ty + i]);
}

// ---------------------------------------------------------------------------
// LayerNorm over D=2048, f32 in -> bf16 out
// ---------------------------------------------------------------------------
__global__ __launch_bounds__(256) void k_ln(const float* __restrict__ x,
                                            const float* __restrict__ w,
                                            const float* __restrict__ b,
                                            unsigned short* __restrict__ out) {
    const int row = blockIdx.x;
    const int tid = threadIdx.x;
    const float* xr = x + (size_t)row * DD;
    float vals[8], sum = 0.f, sq = 0.f;
#pragma unroll
    for (int i = 0; i < 8; ++i) {
        float v = xr[tid + i * 256];
        vals[i] = v; sum += v; sq += v * v;
    }
    __shared__ float s1[256], s2[256];
    s1[tid] = sum; s2[tid] = sq;
    __syncthreads();
    for (int st = 128; st > 0; st >>= 1) {
        if (tid < st) { s1[tid] += s1[tid + st]; s2[tid] += s2[tid + st]; }
        __syncthreads();
    }
    const float mean = s1[0] * (1.0f / DD);
    const float var  = s2[0] * (1.0f / DD) - mean * mean;
    const float rstd = rsqrtf(var + 1e-5f);
#pragma unroll
    for (int i = 0; i < 8; ++i) {
        int c = tid + i * 256;
        out[(size_t)row * DD + c] = f2bf((vals[i] - mean) * rstd * w[c] + b[c]);
    }
}

// ---------------------------------------------------------------------------
// Tiled WMMA GEMM: C[M,N] = A[M,K](bf16) * Wt[N,K](bf16)^T + bias
// 128 threads = 4 waves, 64x64 block tile, K-tile 64, double-buffered LDS
// staged by async global->LDS copies (ASYNCcnt).
// act: 0 = none, 1 = exact GELU.  resid (f32) optionally added.
// ---------------------------------------------------------------------------
#define LDW 72   // LDS row stride in ushorts (144B rows keep 16B alignment)

__global__ __launch_bounds__(128) void k_gemm(const unsigned short* __restrict__ A,
                                              const unsigned short* __restrict__ Bt,
                                              const float* __restrict__ bias,
                                              const float* __restrict__ resid,
                                              float* __restrict__ Cf,
                                              unsigned short* __restrict__ Cb,
                                              int M, int N, int K, int act) {
    __shared__ __align__(16) unsigned short As[2][64 * LDW];
    __shared__ __align__(16) unsigned short Bs[2][64 * LDW];
    const int tid  = threadIdx.x;
    const int lane = tid & 31;
    const int w    = tid >> 5;
    const int wm   = (w >> 1) * 32;
    const int wn   = (w & 1) * 32;
    const int bm   = blockIdx.y * 64;
    const int bn   = blockIdx.x * 64;
    const int half = lane >> 4;
    const int l16  = lane & 15;

    v8f acc[2][2];
#pragma unroll
    for (int i = 0; i < 2; ++i)
#pragma unroll
        for (int j = 0; j < 2; ++j) acc[i][j] = zero8();

    // stage one 64(M/N) x 64(K) tile pair into LDS buffer p (8 async copies)
    auto load_tiles = [&](int k0, int p) {
#pragma unroll
        for (int c = 0; c < 4; ++c) {
            const int idx = tid + c * 128;          // 0..511
            const int r = idx >> 3;                 // 0..63
            const int s = (idx & 7) << 3;           // 0..56 step 8
            async_cp16(A + (size_t)(bm + r) * K + k0 + s, &As[p][r * LDW + s]);
            async_cp16(Bt + (size_t)(bn + r) * K + k0 + s, &Bs[p][r * LDW + s]);
        }
    };

    load_tiles(0, 0);
    int p = 0;
    for (int k0 = 0; k0 < K; k0 += 64) {
        const bool more = (k0 + 64) < K;
        if (more) {
            load_tiles(k0 + 64, p ^ 1);
            wait_async<8>();     // previous tile's 8 copies done (per wave)
        } else {
            wait_async<0>();
        }
        __syncthreads();         // all waves' copies for buffer p landed

#pragma unroll
        for (int ks = 0; ks < 2; ++ks) {
            FragB16 af[2], bfr[2];
#pragma unroll
            for (int mf = 0; mf < 2; ++mf) {
                const unsigned short* rp = &As[p][(wm + mf * 16 + l16) * LDW + ks * 32];
                af[mf].q[0] = *(const uint4*)(rp + half * 16);
                af[mf].q[1] = *(const uint4*)(rp + half * 16 + 8);
            }
#pragma unroll
            for (int nf = 0; nf < 2; ++nf) {
                const unsigned short* rp = &Bs[p][(wn + nf * 16 + l16) * LDW + ks * 32];
                bfr[nf].q[0] = *(const uint4*)(rp + half * 16);
                bfr[nf].q[1] = *(const uint4*)(rp + half * 16 + 8);
            }
#pragma unroll
            for (int mf = 0; mf < 2; ++mf)
#pragma unroll
                for (int nf = 0; nf < 2; ++nf)
                    acc[mf][nf] = wmma_bf16(af[mf], bfr[nf], acc[mf][nf]);
        }
        __syncthreads();         // compute done before buffer p is re-filled
        p ^= 1;
    }

    // epilogue
#pragma unroll
    for (int mf = 0; mf < 2; ++mf)
#pragma unroll
        for (int nf = 0; nf < 2; ++nf) {
            const int col = bn + wn + nf * 16 + l16;
            const float bv = bias ? bias[col] : 0.0f;
#pragma unroll
            for (int i = 0; i < 8; ++i) {
                const int row = bm + wm + mf * 16 + i + half * 8;
                float v = acc[mf][nf][i] + bv;
                if (act == 1) v = 0.5f * v * (1.0f + erff(v * 0.70710678118f));
                if (resid) v += resid[(size_t)row * N + col];
                if (Cf) Cf[(size_t)row * N + col] = v;
                if (Cb) Cb[(size_t)row * N + col] = f2bf(v);
            }
        }
}

// ---------------------------------------------------------------------------
// RoPE helpers / prep kernels
// ---------------------------------------------------------------------------
__device__ __forceinline__ void rope_cs(int s, int d, float& c, float& sn) {
    const int i2 = d & 31;
    const float inv = powf(10000.0f, -((float)(2 * i2)) * (1.0f / 64.0f));
    sincosf((float)s * inv, &sn, &c);
}

// q bf16 [bs][h*192+c] -> qcat bf16 [b][h][s][192] with roped tail
__global__ __launch_bounds__(256) void k_rope_q(const unsigned short* __restrict__ q,
                                                unsigned short* __restrict__ qcat) {
    size_t idx = (size_t)blockIdx.x * 256 + threadIdx.x;
    if (idx >= (size_t)MM * HH * HDQ) return;
    const int c = (int)(idx % HDQ);
    size_t t = idx / HDQ;
    const int h = (int)(t % HH);
    const size_t bs = t / HH;
    const int s = (int)(bs % SS);
    const int b = (int)(bs / SS);
    const unsigned short* qrow = q + (bs * HH + h) * HDQ;
    float v;
    if (c < DNN) {
        v = bf2f(qrow[c]);
    } else {
        const int d = c - DNN;
        const float t0 = bf2f(qrow[DNN + d]);
        const float rot = (d < 32) ? -bf2f(qrow[DNN + d + 32]) : bf2f(qrow[DNN + d - 32]);
        float cs, sn; rope_cs(s, d, cs, sn);
        v = t0 * cs + rot * sn;
    }
    qcat[((size_t)(b * HH + h) * SS + s) * HDQ + c] = f2bf(v);
}

// ckr f32 [bs][576] -> new_kv f32 (ckv | roped kr), ckv bf16, kr bf16
__global__ __launch_bounds__(256) void k_ckv_kr(const float* __restrict__ ckr,
                                                float* __restrict__ nkv,
                                                unsigned short* __restrict__ ckvb,
                                                unsigned short* __restrict__ krb) {
    size_t idx = (size_t)blockIdx.x * 256 + threadIdx.x;
    if (idx >= (size_t)MM * (LATT + DRR)) return;
    const int c = (int)(idx % (LATT + DRR));
    const size_t bs = idx / (LATT + DRR);
    const int s = (int)(bs % SS);
    const float* row = ckr + bs * (LATT + DRR);
    if (c < LATT) {
        const float v = row[c];
        nkv[bs * (LATT + DRR) + c] = v;
        ckvb[bs * LATT + c] = f2bf(v);
    } else {
        const int d = c - LATT;
        const float t0 = row[LATT + d];
        const float rot = (d < 32) ? -row[LATT + d + 32] : row[LATT + d - 32];
        float cs, sn; rope_cs(s, d, cs, sn);
        const float v = t0 * cs + rot * sn;
        nkv[bs * (LATT + DRR) + c] = v;
        krb[bs * DRR + d] = f2bf(v);
    }
}

// knv bf16 [bs][h*256+c] + kr bf16 -> kcat [b][h][s][192], vT [b][h][dv][s]
__global__ __launch_bounds__(256) void k_build(const unsigned short* __restrict__ knv,
                                               const unsigned short* __restrict__ krb,
                                               unsigned short* __restrict__ kcat,
                                               unsigned short* __restrict__ vT) {
    size_t idx = (size_t)blockIdx.x * 256 + threadIdx.x;
    if (idx >= (size_t)MM * HH * (HDQ + DVV)) return;
    const int c = (int)(idx % (HDQ + DVV));
    size_t t = idx / (HDQ + DVV);
    const int h = (int)(t % HH);
    const size_t bs = t / HH;
    const int s = (int)(bs % SS);
    const int b = (int)(bs / SS);
    if (c < HDQ) {
        unsigned short v = (c < DNN) ? knv[(bs * HH + h) * (DNN + DVV) + c]
                                     : krb[bs * DRR + (c - DNN)];
        kcat[((size_t)(b * HH + h) * SS + s) * HDQ + c] = v;
    } else {
        const int dv = c - HDQ;
        vT[((size_t)(b * HH + h) * DVV + dv) * SS + s] =
            knv[(bs * HH + h) * (DNN + DVV) + DNN + dv];
    }
}

// ---------------------------------------------------------------------------
// Flash attention: one wave per (b,h,16-row q tile). 32-key tiles.
// scores = qcat(16x192) @ kcat^T via 12 WMMA; P@V via 8 WMMA against vT.
// ---------------------------------------------------------------------------
#define LDP 40

__global__ __launch_bounds__(128) void k_attn(const unsigned short* __restrict__ qcat,
                                              const unsigned short* __restrict__ kcat,
                                              const unsigned short* __restrict__ vT,
                                              unsigned short* __restrict__ ob) {
    __shared__ __align__(16) unsigned short Pl[4][16 * LDP];
    const int tid = threadIdx.x, lane = tid & 31, w = tid >> 5;
    const int half = lane >> 4, l16 = lane & 15;
    const int wave = blockIdx.x * 4 + w;
    const int QT = SS / 16;               // 128 q tiles
    const int qt = wave % QT;
    const int bh = wave / QT;             // b*16 + h
    const unsigned short* qbase = qcat + ((size_t)bh * SS + qt * 16) * HDQ;
    const unsigned short* kbase = kcat + (size_t)bh * SS * HDQ;
    const unsigned short* vbase = vT + (size_t)bh * DVV * SS;

    // Q fragments (A layout), loaded once
    FragB16 qa[6];
#pragma unroll
    for (int f = 0; f < 6; ++f) {
        const unsigned short* rp = qbase + (size_t)l16 * HDQ + f * 32;
        qa[f].q[0] = *(const uint4*)(rp + half * 16);
        qa[f].q[1] = *(const uint4*)(rp + half * 16 + 8);
    }

    v8f O[8];
#pragma unroll
    for (int j = 0; j < 8; ++j) O[j] = zero8();
    float mrow[8], lrow[8];
#pragma unroll
    for (int i = 0; i < 8; ++i) { mrow[i] = -1e30f; lrow[i] = 0.0f; }

    const int nkt = (qt * 16 + 16 + 31) >> 5;
    for (int kt = 0; kt < nkt; ++kt) {
        v8f s0 = zero8(), s1 = zero8();
#pragma unroll
        for (int f = 0; f < 6; ++f) {
            FragB16 b0, b1;
            const unsigned short* k0p =
                kbase + (size_t)(kt * 32 + l16) * HDQ + f * 32 + half * 16;
            b0.q[0] = *(const uint4*)(k0p);
            b0.q[1] = *(const uint4*)(k0p + 8);
            const unsigned short* k1p = k0p + 16 * HDQ;
            b1.q[0] = *(const uint4*)(k1p);
            b1.q[1] = *(const uint4*)(k1p + 8);
            s0 = wmma_bf16(qa[f], b0, s0);
            s1 = wmma_bf16(qa[f], b1, s1);
        }
        const int qrow = qt * 16 + half * 8;
        const int key0 = kt * 32 + l16, key1 = key0 + 16;
        float tmax[8];
#pragma unroll
        for (int i = 0; i < 8; ++i) {
            float a = s0[i] * ATTN_SCALE; if (key0 > qrow + i) a = -1e30f; s0[i] = a;
            float b = s1[i] * ATTN_SCALE; if (key1 > qrow + i) b = -1e30f; s1[i] = b;
            float t = fmaxf(a, b);
            t = fmaxf(t, __shfl_xor(t, 1));
            t = fmaxf(t, __shfl_xor(t, 2));
            t = fmaxf(t, __shfl_xor(t, 4));
            t = fmaxf(t, __shfl_xor(t, 8));
            tmax[i] = t;
        }
#pragma unroll
        for (int i = 0; i < 8; ++i) {
            const float mn = fmaxf(mrow[i], tmax[i]);
            const float sc = __expf(mrow[i] - mn);
            const float p0 = __expf(s0[i] - mn);
            const float p1 = __expf(s1[i] - mn);
            float ts = p0 + p1;
            ts += __shfl_xor(ts, 1);
            ts += __shfl_xor(ts, 2);
            ts += __shfl_xor(ts, 4);
            ts += __shfl_xor(ts, 8);
            lrow[i] = lrow[i] * sc + ts;
            mrow[i] = mn;
            Pl[w][(i + half * 8) * LDP + l16]      = f2bf(p0);
            Pl[w][(i + half * 8) * LDP + 16 + l16] = f2bf(p1);
#pragma unroll
            for (int j = 0; j < 8; ++j) O[j][i] *= sc;
        }
        // C-layout -> A-layout through LDS (wave-private; DS ops are in-order)
        FragB16 pa;
        const unsigned short* pp = Pl[w] + l16 * LDP;
        pa.q[0] = *(const uint4*)(pp + half * 16);
        pa.q[1] = *(const uint4*)(pp + half * 16 + 8);
#pragma unroll
        for (int j = 0; j < 8; ++j) {
            FragB16 vb;
            const unsigned short* vp =
                vbase + (size_t)(j * 16 + l16) * SS + kt * 32 + half * 16;
            vb.q[0] = *(const uint4*)(vp);
            vb.q[1] = *(const uint4*)(vp + 8);
            O[j] = wmma_bf16(pa, vb, O[j]);
        }
    }

    const int b = bh >> 4, h = bh & 15;
#pragma unroll
    for (int j = 0; j < 8; ++j)
#pragma unroll
        for (int i = 0; i < 8; ++i) {
            const int s = qt * 16 + i + half * 8;
            const int dv = j * 16 + l16;
            ob[((size_t)(b * SS + s)) * (HH * DVV) + h * DVV + dv] =
                f2bf(O[j][i] / lrow[i]);
        }
}

// ---------------------------------------------------------------------------
// Host-side launch
// ---------------------------------------------------------------------------
extern "C" void kernel_launch(void* const* d_in, const int* in_sizes, int n_in,
                              void* d_out, int out_size, void* d_ws, size_t ws_size,
                              hipStream_t stream) {
    (void)in_sizes; (void)n_in; (void)out_size; (void)ws_size;
    const float* x    = (const float*)d_in[0];
    const float* ln1w = (const float*)d_in[1];
    const float* ln1b = (const float*)d_in[2];
    const float* ln2w = (const float*)d_in[3];
    const float* ln2b = (const float*)d_in[4];
    const float* wq   = (const float*)d_in[5];
    const float* bq   = (const float*)d_in[6];
    const float* wdkv = (const float*)d_in[7];
    const float* bdkv = (const float*)d_in[8];
    const float* wkv  = (const float*)d_in[9];
    const float* bkv  = (const float*)d_in[10];
    const float* wo   = (const float*)d_in[11];
    const float* bo   = (const float*)d_in[12];
    const float* w1   = (const float*)d_in[13];
    const float* b1   = (const float*)d_in[14];
    const float* w2   = (const float*)d_in[15];
    const float* b2   = (const float*)d_in[16];

    float* outx  = (float*)d_out;
    float* outkv = outx + (size_t)MM * DD;

    char* base = (char*)d_ws;
    size_t off = 0;
    auto alloc = [&](size_t bytes) -> void* {
        void* p = base + off;
        off = (off + bytes + 255) & ~(size_t)255;
        return p;
    };
    typedef unsigned short us;
    us* hb     = (us*)alloc((size_t)MM * DD * 2);
    us* h2b    = (us*)alloc((size_t)MM * DD * 2);
    us* wq_t   = (us*)alloc((size_t)(HH * HDQ) * DD * 2);
    us* wdkv_t = (us*)alloc((size_t)(LATT + DRR) * DD * 2);
    us* wkv_t  = (us*)alloc((size_t)(HH * (DNN + DVV)) * LATT * 2);
    us* wo_t   = (us*)alloc((size_t)DD * (HH * DVV) * 2);
    us* w1_t   = (us*)alloc((size_t)DFFF * DD * 2);
    us* w2_t   = (us*)alloc((size_t)DD * DFFF * 2);
    us* qb     = (us*)alloc((size_t)MM * HH * HDQ * 2);
    us* qcat   = (us*)alloc((size_t)MM * HH * HDQ * 2);
    float* ckr = (float*)alloc((size_t)MM * (LATT + DRR) * 4);
    us* ckvb   = (us*)alloc((size_t)MM * LATT * 2);
    us* krb    = (us*)alloc((size_t)MM * DRR * 2);
    us* knvb   = (us*)alloc((size_t)MM * HH * (DNN + DVV) * 2);
    us* kcat   = (us*)alloc((size_t)MM * HH * HDQ * 2);
    us* vTb    = (us*)alloc((size_t)BB * HH * DVV * SS * 2);
    us* obuf   = (us*)alloc((size_t)MM * HH * DVV * 2);
    float* x1  = (float*)alloc((size_t)MM * DD * 4);
    us* ff1    = (us*)alloc((size_t)MM * DFFF * 2);

    // 1) weight transposes -> bf16 [N][K]
    k_wt<<<dim3((HH * HDQ) / 32, DD / 32), 256, 0, stream>>>(wq, wq_t, DD, HH * HDQ);
    k_wt<<<dim3((LATT + DRR) / 32, DD / 32), 256, 0, stream>>>(wdkv, wdkv_t, DD, LATT + DRR);
    k_wt<<<dim3((HH * (DNN + DVV)) / 32, LATT / 32), 256, 0, stream>>>(wkv, wkv_t, LATT, HH * (DNN + DVV));
    k_wt<<<dim3(DD / 32, (HH * DVV) / 32), 256, 0, stream>>>(wo, wo_t, HH * DVV, DD);
    k_wt<<<dim3(DFFF / 32, DD / 32), 256, 0, stream>>>(w1, w1_t, DD, DFFF);
    k_wt<<<dim3(DD / 32, DFFF / 32), 256, 0, stream>>>(w2, w2_t, DFFF, DD);

    // 2) LN1
    k_ln<<<MM, 256, 0, stream>>>(x, ln1w, ln1b, hb);

    // 3) q = h @ wq + bq  (bf16 out)
    k_gemm<<<dim3((HH * HDQ) / 64, MM / 64), 128, 0, stream>>>(
        hb, wq_t, bq, nullptr, nullptr, qb, MM, HH * HDQ, DD, 0);
    // 4) ckr = h @ wdkv + bdkv  (f32 out)
    k_gemm<<<dim3((LATT + DRR) / 64, MM / 64), 128, 0, stream>>>(
        hb, wdkv_t, bdkv, nullptr, ckr, nullptr, MM, LATT + DRR, DD, 0);
    // 5) split ckr: new_kv (ckv | roped kr), ckv bf16, kr bf16
    k_ckv_kr<<<(MM * (LATT + DRR) + 255) / 256, 256, 0, stream>>>(ckr, outkv, ckvb, krb);
    // 6) knv = ckv @ wkv + bkv  (bf16 out)
    k_gemm<<<dim3((HH * (DNN + DVV)) / 64, MM / 64), 128, 0, stream>>>(
        ckvb, wkv_t, bkv, nullptr, nullptr, knvb, MM, HH * (DNN + DVV), LATT, 0);
    // 7) rope q -> qcat ; 8) assemble kcat / vT
    k_rope_q<<<((size_t)MM * HH * HDQ + 255) / 256, 256, 0, stream>>>(qb, qcat);
    k_build<<<((size_t)MM * HH * (HDQ + DVV) + 255) / 256, 256, 0, stream>>>(knvb, krb, kcat, vTb);
    // 9) flash attention
    k_attn<<<(BB * HH * (SS / 16)) / 4, 128, 0, stream>>>(qcat, kcat, vTb, obuf);
    // 10) x1 = x + o @ wo + bo
    k_gemm<<<dim3(DD / 64, MM / 64), 128, 0, stream>>>(
        obuf, wo_t, bo, x, x1, nullptr, MM, DD, HH * DVV, 0);
    // 11) LN2
    k_ln<<<MM, 256, 0, stream>>>(x1, ln2w, ln2b, h2b);
    // 12) ff1 = gelu(h2 @ w1 + b1)  (bf16 out)
    k_gemm<<<dim3(DFFF / 64, MM / 64), 128, 0, stream>>>(
        h2b, w1_t, b1, nullptr, nullptr, ff1, MM, DFFF, DD, 1);
    // 13) out x = x1 + ff1 @ w2 + b2
    k_gemm<<<dim3(DD / 64, MM / 64), 128, 0, stream>>>(
        ff1, w2_t, b2, x1, outx, nullptr, MM, DD, DFFF, 0);
}
